// UnifiedQwen3VLAttention_13932873908593
// MI455X (gfx1250) — compile-verified, matrix operand order checked by reading, not run on hardware
//
#include <hip/hip_runtime.h>
#include <hip/hip_bf16.h>

// MI455X (gfx1250, wave32) attention block:
//   y = softmax( (xWq)(xWk)^T / sqrt(128) + causal ) (xWv) Wo
//
// Design for this chip:
//  * All matmuls on v_wmma_f32_16x16x32_f16 (f32 accumulate).
//  * One-time f32->f16 conversion of hidden + transposed weights (~0.5GB of
//    streaming = ~20us at 23.3 TB/s; trivially amortized vs ~0.8 TFLOP compute).
//  * GEMM tiles staged by the Tensor Data Mover (tensor_load_to_lds, double
//    buffered, s_wait_tensorcnt) with TDM padding -> 40-half LDS row stride.
//    Cooperative b128 fallback if the builtin is unavailable.
//  * Flash attention: K/V blocks cooperatively staged to LDS (uniform key loop
//    over the whole 64-row q tile; trailing masked blocks are no-ops under
//    online softmax), removing the scalar global_load_u16 storm of round 1.

typedef _Float16 v16h __attribute__((ext_vector_type(16)));
typedef _Float16 v8h  __attribute__((ext_vector_type(8)));
typedef float    v8f  __attribute__((ext_vector_type(8)));
typedef unsigned int u32x4 __attribute__((ext_vector_type(4)));
typedef int i32x4 __attribute__((ext_vector_type(4)));
typedef int i32x8 __attribute__((ext_vector_type(8)));

#define WMMA_F16(a, b, c) \
    __builtin_amdgcn_wmma_f32_16x16x32_f16(false, (a), false, (b), (short)0, (c), false, false)

#define V8F_ZERO (v8f){0.f,0.f,0.f,0.f,0.f,0.f,0.f,0.f}

#if defined(__has_builtin)
#if __has_builtin(__builtin_amdgcn_tensor_load_to_lds) && \
    __has_builtin(__builtin_amdgcn_s_wait_tensorcnt)
#define HAVE_TDM 1
#endif
#endif
#ifndef HAVE_TDM
#define HAVE_TDM 0
#endif

#if HAVE_TDM
// 2D f16 tile DMA: global (row-major, stride_elems between rows) -> LDS with
// +4-DWORD padding every 16 DWORDs (tile_w must be 32 halves -> LDS row
// stride = 40 halves). D# layout per CDNA5 ISA 8.3/8.4.
__device__ __forceinline__ void tdm_load_2d_f16_pad40(unsigned lds_off, const void* gsrc,
                                                      unsigned tile_w, unsigned tile_h,
                                                      unsigned stride_elems,
                                                      unsigned tensor_w, unsigned tensor_h) {
    unsigned long long ga = (unsigned long long)(size_t)gsrc;
    u32x4 g0;
    g0[0] = 1u;                                                 // count=1, user D#
    g0[1] = lds_off;                                            // lds_addr (bytes)
    g0[2] = (unsigned)ga;                                       // global_addr[31:0]
    g0[3] = (unsigned)((ga >> 32) & 0x01ffffffu) | (2u << 30);  // [56:32] | type=2
    i32x8 g1;
    // data_size=1 (2B) | pad_enable | pad_interval=3 (16 dw) | pad_amount=3 (4 dw)
    g1[0] = (int)((1u << 16) | (1u << 20) | (3u << 22) | (3u << 25));
    g1[1] = (int)((tensor_w & 0xffffu) << 16);                              // dim0[15:0]
    g1[2] = (int)(((tensor_w >> 16) & 0xffffu) | ((tensor_h & 0xffffu) << 16));
    g1[3] = (int)(((tensor_h >> 16) & 0xffffu) | ((tile_w & 0xffffu) << 16));
    g1[4] = (int)(tile_h & 0xffffu);                                        // tile_dim1
    g1[5] = (int)stride_elems;                                              // dim0_stride
    g1[6] = 0;
    g1[7] = 0;
    i32x4 z4 = {0, 0, 0, 0};
#if __clang_major__ >= 23
    i32x8 z8 = {0, 0, 0, 0, 0, 0, 0, 0};
    __builtin_amdgcn_tensor_load_to_lds(g0, g1, z4, z4, z8, 0);
#else
    __builtin_amdgcn_tensor_load_to_lds(g0, g1, z4, z4, 0);
#endif
}
#endif

// ---------------------------------------------------------------------------
// Elementwise f32 -> f16 (8 elements/thread, b128 in / b128 out)
// ---------------------------------------------------------------------------
__global__ __launch_bounds__(256) void convert_f32_f16(const float* __restrict__ X,
                                                       _Float16* __restrict__ Y, size_t n) {
    size_t base = ((size_t)blockIdx.x * blockDim.x + threadIdx.x) * 8;
    if (base + 8 > n) return;
    float4 a = *(const float4*)(X + base);
    float4 b = *(const float4*)(X + base + 4);
    v8h o;
    o[0] = (_Float16)a.x; o[1] = (_Float16)a.y; o[2] = (_Float16)a.z; o[3] = (_Float16)a.w;
    o[4] = (_Float16)b.x; o[5] = (_Float16)b.y; o[6] = (_Float16)b.z; o[7] = (_Float16)b.w;
    *(v8h*)(Y + base) = o;
}

// ---------------------------------------------------------------------------
// Transpose + convert: W[K,N] f32 -> Wt[N,K] f16 (32x32 LDS tiles)
// ---------------------------------------------------------------------------
__global__ __launch_bounds__(256) void transpose_f32_f16(const float* __restrict__ W,
                                                         _Float16* __restrict__ Wt,
                                                         int Kd, int Nd) {
    __shared__ _Float16 t[32][33];
    const int bx = blockIdx.x * 32;  // N offset
    const int by = blockIdx.y * 32;  // K offset
    const int tx = threadIdx.x & 31;
    const int ty = threadIdx.x >> 5;  // 0..7
#pragma unroll
    for (int i = 0; i < 4; ++i)
        t[ty + 8 * i][tx] = (_Float16)W[(size_t)(by + ty + 8 * i) * Nd + bx + tx];
    __syncthreads();
#pragma unroll
    for (int i = 0; i < 4; ++i)
        Wt[(size_t)(bx + ty + 8 * i) * Kd + by + tx] = t[tx][ty + 8 * i];
}

// ---------------------------------------------------------------------------
// WMMA GEMM (NT): C[M,N] = A[M,K](f16) * Bt[N,K](f16)^T
// 256 threads = 8 waves; block tile 128x128; wave tile 32x64 (2x4 WMMA accums).
// Tiles staged via TDM (double-buffered) or cooperative b128 loads (fallback).
// ---------------------------------------------------------------------------
template <typename TOUT>
__global__ __launch_bounds__(256) void gemm_nt_f16(const _Float16* __restrict__ A,
                                                   const _Float16* __restrict__ Bt,
                                                   TOUT* __restrict__ C,
                                                   int M, int N, int K) {
    __shared__ __align__(16) _Float16 As[2][128][40];
    __shared__ __align__(16) _Float16 Bs[2][128][40];

    const int tid  = threadIdx.x;
    const int lane = tid & 31;
    const int w    = tid >> 5;
    const int wm   = (w & 3) * 32;
    const int wn   = (w >> 2) * 64;
    const int bm   = blockIdx.y * 128;
    const int bn   = blockIdx.x * 128;
    const int koff = (lane >> 4) * 8;
    const int hi8  = (lane >> 4) * 8;

    v8f acc[2][4];
#pragma unroll
    for (int i = 0; i < 2; ++i)
#pragma unroll
        for (int j = 0; j < 4; ++j) acc[i][j] = V8F_ZERO;

    const _Float16* Ab = A + (size_t)bm * K;
    const _Float16* Bb = Bt + (size_t)bn * K;

    auto mma_tile = [&](int buf) {
        v16h af[2];
#pragma unroll
        for (int fm = 0; fm < 2; ++fm) {
            const int row = wm + fm * 16 + (lane & 15);
#pragma unroll
            for (int j = 0; j < 8; ++j) {
                af[fm][j]     = As[buf][row][koff + j];
                af[fm][j + 8] = As[buf][row][koff + 16 + j];
            }
        }
#pragma unroll
        for (int fn = 0; fn < 4; ++fn) {
            v16h bf;
            const int col = wn + fn * 16 + (lane & 15);
#pragma unroll
            for (int j = 0; j < 8; ++j) {
                bf[j]     = Bs[buf][col][koff + j];
                bf[j + 8] = Bs[buf][col][koff + 16 + j];
            }
#pragma unroll
            for (int fm = 0; fm < 2; ++fm)
                acc[fm][fn] = WMMA_F16(af[fm], bf, acc[fm][fn]);
        }
    };

#if HAVE_TDM
    const unsigned as0 = (unsigned)(size_t)&As[0][0][0];
    const unsigned bs0 = (unsigned)(size_t)&Bs[0][0][0];
    const unsigned bufbytes = 128u * 40u * (unsigned)sizeof(_Float16);
    if (w == 0) {
        tdm_load_2d_f16_pad40(as0, Ab, 32, 128, (unsigned)K, (unsigned)K, 128);
        tdm_load_2d_f16_pad40(bs0, Bb, 32, 128, (unsigned)K, (unsigned)K, 128);
        __builtin_amdgcn_s_wait_tensorcnt(0);
    }
    __syncthreads();
    int buf = 0;
    for (int k0 = 0; k0 < K; k0 += 32) {
        const bool more = (k0 + 32) < K;
        if (w == 0 && more) {  // prefetch next tiles into the other buffer
            tdm_load_2d_f16_pad40(as0 + (buf ^ 1) * bufbytes, Ab + k0 + 32, 32, 128,
                                  (unsigned)K, (unsigned)K, 128);
            tdm_load_2d_f16_pad40(bs0 + (buf ^ 1) * bufbytes, Bb + k0 + 32, 32, 128,
                                  (unsigned)K, (unsigned)K, 128);
        }
        mma_tile(buf);
        if (w == 0 && more) __builtin_amdgcn_s_wait_tensorcnt(0);
        __syncthreads();
        buf ^= 1;
    }
#else
    const int lr = tid >> 1;        // 0..127
    const int lc = (tid & 1) * 16;  // 0 or 16
    for (int k0 = 0; k0 < K; k0 += 32) {
        *(v8h*)&As[0][lr][lc]     = *(const v8h*)(Ab + (size_t)lr * K + k0 + lc);
        *(v8h*)&As[0][lr][lc + 8] = *(const v8h*)(Ab + (size_t)lr * K + k0 + lc + 8);
        *(v8h*)&Bs[0][lr][lc]     = *(const v8h*)(Bb + (size_t)lr * K + k0 + lc);
        *(v8h*)&Bs[0][lr][lc + 8] = *(const v8h*)(Bb + (size_t)lr * K + k0 + lc + 8);
        __syncthreads();
        mma_tile(0);
        __syncthreads();
    }
#endif

    // epilogue: C frag — VGPR r: lanes 0-15 => M=r, lanes 16-31 => M=r+8
#pragma unroll
    for (int fm = 0; fm < 2; ++fm)
#pragma unroll
        for (int fn = 0; fn < 4; ++fn) {
            const int col = bn + wn + fn * 16 + (lane & 15);
#pragma unroll
            for (int r = 0; r < 8; ++r) {
                const int row = bm + wm + fm * 16 + r + hi8;
                C[(size_t)row * N + col] = (TOUT)acc[fm][fn][r];
            }
        }
}

// ---------------------------------------------------------------------------
// Causal flash attention, f16 Q/K/V in [B,S,H,HD], f32 online softmax.
// 128 threads = 4 waves; q tile = 64 rows (16/wave). Uniform key loop over the
// block's diagonal so K/V 32x128 tiles are staged cooperatively in LDS once
// per block (4x less global traffic than per-wave loading); trailing masked
// blocks contribute exp(-1e9-m)=0 and alpha=1 -> exact no-ops.
// ---------------------------------------------------------------------------
__global__ __launch_bounds__(128) void flash_attn_f16(const _Float16* __restrict__ Q,
                                                      const _Float16* __restrict__ Kf,
                                                      const _Float16* __restrict__ Vf,
                                                      _Float16* __restrict__ O,
                                                      int Bn, int S, int Hn, int HD) {
    __shared__ __align__(16) _Float16 Ks[32][136];
    __shared__ __align__(16) _Float16 Vs[32][136];
    __shared__ __align__(16) _Float16 Ps[4][16][40];

    const int tid  = threadIdx.x;
    const int lane = tid & 31;
    const int w    = tid >> 5;
    const int bb   = blockIdx.y / Hn;
    const int hh   = blockIdx.y % Hn;
    const int qblk = blockIdx.x * 64;
    const int qbase = qblk + w * 16;
    const int col  = lane & 15;
    const int koff = (lane >> 4) * 8;
    const int hi8  = (lane >> 4) * 8;
    const float scale = 0.08838834764831845f;  // 1/sqrt(128)

    // Q fragments for this wave's 16 rows (HD=128 -> 4 chunks of K=32)
    v16h aq[4];
    {
        const int qrow = qbase + (lane & 15);
        const _Float16* qp = Q + ((size_t)(bb * S + qrow) * Hn + hh) * HD;
#pragma unroll
        for (int c = 0; c < 4; ++c) {
            v8h lo = *(const v8h*)(qp + c * 32 + koff);
            v8h hi = *(const v8h*)(qp + c * 32 + koff + 16);
#pragma unroll
            for (int j = 0; j < 8; ++j) { aq[c][j] = lo[j]; aq[c][j + 8] = hi[j]; }
        }
    }

    v8f o[8];
#pragma unroll
    for (int i = 0; i < 8; ++i) o[i] = V8F_ZERO;
    float mrow[8], lrow[8];
#pragma unroll
    for (int r = 0; r < 8; ++r) { mrow[r] = -3.0e38f; lrow[r] = 0.f; }

    // cooperative staging assignment: thread -> (key row, 32-half d span)
    const int skey = tid >> 2;
    const int sd   = (tid & 3) * 32;

    const int kb_end = qblk + 63;  // block-uniform causal bound
    for (int kb = 0; kb <= kb_end; kb += 32) {
        {   // stage K/V 32x128 tiles (b128 in / b128 to LDS)
            const size_t rowoff = ((size_t)(bb * S + kb + skey) * Hn + hh) * HD + sd;
            const _Float16* kp = Kf + rowoff;
            const _Float16* vp = Vf + rowoff;
#pragma unroll
            for (int c = 0; c < 4; ++c) {
                *(v8h*)&Ks[skey][sd + c * 8] = *(const v8h*)(kp + c * 8);
                *(v8h*)&Vs[skey][sd + c * 8] = *(const v8h*)(vp + c * 8);
            }
        }
        __syncthreads();

        // ---- scores for two 16-key halves
        v8f sc[2];
#pragma unroll
        for (int h2 = 0; h2 < 2; ++h2) {
            const int kl = h2 * 16 + col;  // key row in LDS tile
            v8f s = V8F_ZERO;
#pragma unroll
            for (int c = 0; c < 4; ++c) {
                v16h bf;
                v8h lo = *(const v8h*)&Ks[kl][c * 32 + koff];
                v8h hi = *(const v8h*)&Ks[kl][c * 32 + koff + 16];
#pragma unroll
                for (int j = 0; j < 8; ++j) { bf[j] = lo[j]; bf[j + 8] = hi[j]; }
                s = WMMA_F16(aq[c], bf, s);
            }
            sc[h2] = s;
        }

        // ---- online softmax (rows live across 16-lane groups)
#pragma unroll
        for (int r = 0; r < 8; ++r) {
            const int qrow = qbase + r + hi8;
            float v0 = (kb + col <= qrow)      ? sc[0][r] * scale : -1.0e9f;
            float v1 = (kb + 16 + col <= qrow) ? sc[1][r] * scale : -1.0e9f;
            float mx = fmaxf(v0, v1);
#pragma unroll
            for (int off = 1; off < 16; off <<= 1)
                mx = fmaxf(mx, __shfl_xor(mx, off, 32));
            const float mn = fmaxf(mrow[r], mx);
            const float p0 = __expf(v0 - mn);
            const float p1 = __expf(v1 - mn);
            float rs = p0 + p1;
#pragma unroll
            for (int off = 1; off < 16; off <<= 1)
                rs += __shfl_xor(rs, off, 32);
            const float alpha = __expf(mrow[r] - mn);
            lrow[r] = lrow[r] * alpha + rs;
            mrow[r] = mn;
            Ps[w][r + hi8][col]      = (_Float16)p0;
            Ps[w][r + hi8][col + 16] = (_Float16)p1;
#pragma unroll
            for (int fo = 0; fo < 8; ++fo) o[fo][r] *= alpha;
        }

        // ---- P (C-layout) -> A-layout via wave-private LDS scratch
        v16h pf;
        {
            const int row = lane & 15;
#pragma unroll
            for (int j = 0; j < 8; ++j) {
                pf[j]     = Ps[w][row][koff + j];
                pf[j + 8] = Ps[w][row][koff + 16 + j];
            }
        }

        // ---- O += P * V (V fragments from LDS)
#pragma unroll
        for (int fo = 0; fo < 8; ++fo) {
            v16h vf;
            const int d = fo * 16 + col;
#pragma unroll
            for (int j = 0; j < 8; ++j) {
                vf[j]     = Vs[koff + j][d];
                vf[j + 8] = Vs[koff + 16 + j][d];
            }
            o[fo] = WMMA_F16(pf, vf, o[fo]);
        }
        __syncthreads();  // protect Ks/Vs before next staging
    }

    // ---- normalize and write attn-out (f16, [B,S,H,HD])
#pragma unroll
    for (int r = 0; r < 8; ++r) {
        const float inv = 1.0f / lrow[r];
        const int qrow = qbase + r + hi8;
#pragma unroll
        for (int fo = 0; fo < 8; ++fo) {
            const int d = fo * 16 + col;
            O[((size_t)(bb * S + qrow) * Hn + hh) * HD + d] = (_Float16)(o[fo][r] * inv);
        }
    }
}

// ---------------------------------------------------------------------------
extern "C" void kernel_launch(void* const* d_in, const int* in_sizes, int n_in,
                              void* d_out, int out_size, void* d_ws, size_t ws_size,
                              hipStream_t stream) {
    const float* hidden = (const float*)d_in[0];
    // d_in[1] attention_mask == triu(-1e9): applied analytically in-kernel.
    const float* Wq = (const float*)d_in[2];
    const float* Wk = (const float*)d_in[3];
    const float* Wv = (const float*)d_in[4];
    const float* Wo = (const float*)d_in[5];
    float* out = (float*)d_out;

    const int Bn = 2, S = 2048, HID = 4096, Hn = 32, HD = 128;
    const int M = Bn * S;                 // 4096
    const size_t per = (size_t)M * HID;   // 16M elements

    // f16 workspace (9 x 32MB = 288MB): h16, 4 transposed weights, q,k,v,ao
    _Float16* h16 = (_Float16*)d_ws;
    _Float16* wqt = h16 + per;
    _Float16* wkt = wqt + per;
    _Float16* wvt = wkt + per;
    _Float16* wot = wvt + per;
    _Float16* q   = wot + per;
    _Float16* k   = q + per;
    _Float16* v   = k + per;
    _Float16* ao  = v + per;

    // 1) one-time conversions / transposes
    convert_f32_f16<<<dim3((unsigned)(per / (256 * 8))), dim3(256), 0, stream>>>(hidden, h16, per);
    dim3 tgrid(HID / 32, HID / 32);
    transpose_f32_f16<<<tgrid, dim3(256), 0, stream>>>(Wq, wqt, HID, HID);
    transpose_f32_f16<<<tgrid, dim3(256), 0, stream>>>(Wk, wkt, HID, HID);
    transpose_f32_f16<<<tgrid, dim3(256), 0, stream>>>(Wv, wvt, HID, HID);
    transpose_f32_f16<<<tgrid, dim3(256), 0, stream>>>(Wo, wot, HID, HID);

    // 2) Q/K/V projections
    dim3 ggrid(HID / 128, M / 128);
    gemm_nt_f16<_Float16><<<ggrid, dim3(256), 0, stream>>>(h16, wqt, q, M, HID, HID);
    gemm_nt_f16<_Float16><<<ggrid, dim3(256), 0, stream>>>(h16, wkt, k, M, HID, HID);
    gemm_nt_f16<_Float16><<<ggrid, dim3(256), 0, stream>>>(h16, wvt, v, M, HID, HID);

    // 3) causal flash attention
    dim3 fgrid(S / 64, Bn * Hn);
    flash_attn_f16<<<fgrid, dim3(128), 0, stream>>>(q, k, v, ao, Bn, S, Hn, HD);

    // 4) output projection (f32 out)
    gemm_nt_f16<float><<<ggrid, dim3(256), 0, stream>>>(ao, wot, out, M, HID, HID);
}
